// HPE_13786845020567
// MI455X (gfx1250) — matrix operand order repaired and checked by reading, not compiled, not run on hardware
//
#include <hip/hip_runtime.h>

typedef __attribute__((ext_vector_type(16))) _Float16 v16h;
typedef __attribute__((ext_vector_type(8)))  float    v8f;

#define BATCH 8
#define HH 128
#define WW 128
#define LL (HH*WW)          // 16384
#define BL (BATCH*LL)       // 131072
#define GTILES 8
#define GWAVES 4            // waves per GEMM/conv3 block

// ---------------- workspace layout (bytes, all 256-aligned) ----------------
static const size_t OFF_POOL = 0;                 // 393216 f32 = 1572864  (dead after conv7)
static const size_t OFF_P1N  = 0;                 // reuse pool: BL f32
static const size_t OFF_P2N  = 524288;            // BL f32
static const size_t OFF_A1   = 1572864;           // BL*32 f16 = 8388608  (dead after conv3)
static const size_t OFF_A2   = OFF_A1 + 8388608;  // BL*32 f16            (dead after cbr1x1)
static const size_t OFF_YG   = OFF_A1;            // reuse a1+a2: BL*64 f16 = 16777216
static const size_t OFF_X3F  = 18350080;          // BL*32 f32 = 16777216
static const size_t OFF_X3H  = 35127296;          // BL*32 f16 = 8388608
static const size_t OFF_XZ   = 43515904;          // BL*128 f32 = 67108864
static const size_t OFF_U16  = 110624768;         // BL*64 f16 = 16777216
static const size_t OFF_UF   = 127401984;         // BL*64 f32 = 33554432 (dead after scan)
static const size_t OFF_XNF  = OFF_UF;            // reuse: BL*32 f32
static const size_t OFF_XNH  = OFF_UF + 16777216; // BL*32 f16
static const size_t OFF_XDBL = 160956416;         // BL*48 f32 = 25165824
static const size_t OFF_W    = 186122240;         // weights region
// weights sub-offsets (from OFF_W)
static const size_t OW_W7   = 0;       // 32*147 f32 (folded)
static const size_t OW_B7   = 18944;   // 32 f32
static const size_t OW_W3   = 19200;   // 9*32*32 f16 (tap-major, folded)
static const size_t OW_B3   = 37888;   // 32 f32
static const size_t OW_W1   = 39936;   // 32*32 f16 (folded)
static const size_t OW_B1   = 40192;   // 32 f32
static const size_t OW_W4   = 42240;   // 32*32 f16 (folded)
static const size_t OW_B4   = 42496;   // 32 f32
static const size_t OW_INW  = 50688;   // 128*32 f16
static const size_t OW_XPW  = 56832;   // 48*64 f16 (rows 34..47 zero)
static const size_t OW_OUTW = 60928;   // 32*64 f16
static const size_t OW_AC   = 65024;   // 64*16 f32 = -exp(A_log)

// ---------------- WMMA fragment helpers (contiguous k = 16*hi + j in both A and B) ----------------
__device__ __forceinline__ v16h wmma_load_a(const _Float16* __restrict__ Wm, int ld,
                                            int m0, int koff, int lane) {
  int m  = m0 + (lane & 15);
  int hi = (lane >> 4) & 1;
  const _Float16* p = Wm + (long)m * ld + koff;
  v16h a;
#pragma unroll
  for (int j = 0; j < 16; ++j) {
    int k = 16 * hi + j;
    a[j] = p[k];
  }
  return a;
}

__device__ __forceinline__ v16h wmma_load_b(const _Float16* __restrict__ X, long n0,
                                            int ld, int koff, int lane) {
  int n  = lane & 15;
  int hi = (lane >> 4) & 1;
  const _Float16* p = X + (n0 + n) * (long)ld + koff + 16 * hi;
  return *(const v16h*)p;   // 32B aligned: ld multiple of 32, base 256-aligned
}

// ---------------- prep: fold BN into weights, convert to f16 ----------------
__global__ void k_prep(const float* w7, const float* b7, const float* g1, const float* be1,
                       const float* m1, const float* v1,
                       const float* w3, const float* b3, const float* g2, const float* be2,
                       const float* m2, const float* v2,
                       const float* w1, const float* b1, const float* g3, const float* be3,
                       const float* m3, const float* v3,
                       const float* in_w, const float* xp_w, const float* out_w, const float* A_log,
                       const float* w4, const float* b4, const float* g4, const float* be4,
                       const float* m4, const float* v4,
                       float* w7f, float* b7f, _Float16* w3h, float* b3f,
                       _Float16* w1h, float* b1f, _Float16* w4h, float* b4f,
                       _Float16* inwh, _Float16* xpwh, _Float16* outwh, float* Ac) {
  int tid = blockIdx.x * blockDim.x + threadIdx.x;
  int nt  = gridDim.x * blockDim.x;
  for (int i = tid; i < 4704; i += nt) {
    int oc = i / 147;
    float s = g1[oc] * rsqrtf(v1[oc] + 1e-5f);
    w7f[i] = w7[i] * s;
  }
  for (int i = tid; i < 32; i += nt) {
    float s1 = g1[i] * rsqrtf(v1[i] + 1e-5f);
    b7f[i] = (b7[i] - m1[i]) * s1 + be1[i];
    float s2 = g2[i] * rsqrtf(v2[i] + 1e-5f);
    b3f[i] = (b3[i] - m2[i]) * s2 + be2[i];
    float s3 = g3[i] * rsqrtf(v3[i] + 1e-5f);
    b1f[i] = (b1[i] - m3[i]) * s3 + be3[i];
    float s4 = g4[i] * rsqrtf(v4[i] + 1e-5f);
    b4f[i] = (b4[i] - m4[i]) * s4 + be4[i];
  }
  for (int i = tid; i < 9216; i += nt) {          // w3h[tap][oc][ic]
    int tap = i >> 10, rem = i & 1023, oc = rem >> 5, ic = rem & 31;
    float s = g2[oc] * rsqrtf(v2[oc] + 1e-5f);
    w3h[i] = (_Float16)(w3[(oc * 32 + ic) * 9 + tap] * s);
  }
  for (int i = tid; i < 1024; i += nt) {
    int oc = i >> 5;
    float s3 = g3[oc] * rsqrtf(v3[oc] + 1e-5f);
    w1h[i] = (_Float16)(w1[i] * s3);
    float s4 = g4[oc] * rsqrtf(v4[oc] + 1e-5f);
    w4h[i] = (_Float16)(w4[i] * s4);
    Ac[i] = -expf(A_log[i]);
  }
  for (int i = tid; i < 4096; i += nt) inwh[i] = (_Float16)in_w[i];
  for (int i = tid; i < 3072; i += nt) {
    int r = i >> 6, k = i & 63;
    xpwh[i] = (r < 34) ? (_Float16)xp_w[r * 64 + k] : (_Float16)0.f;
  }
  for (int i = tid; i < 2048; i += nt) outwh[i] = (_Float16)out_w[i];
}

// ---------------- 2x2 maxpool ----------------
__global__ void k_pool(const float* __restrict__ x, float* __restrict__ out) {
  int i = blockIdx.x * blockDim.x + threadIdx.x;   // over 8*3*128*128
  if (i >= BATCH * 3 * LL) return;
  int xc = i & 127;
  int t  = i >> 7;
  int yr = t & 127;
  t >>= 7;                                          // t = b*3 + c
  const float* src = x + (((size_t)t * 256 + yr * 2) * 256 + xc * 2);
  out[i] = fmaxf(fmaxf(src[0], src[1]), fmaxf(src[256], src[257]));
}

// ---------------- conv7x7 (+folded BN, ReLU), channel-last f16 out ----------------
__global__ void k_conv7(const float* __restrict__ pool, const float* __restrict__ w7f,
                        const float* __restrict__ b7f, _Float16* __restrict__ a1) {
  __shared__ float sw[32 * 147];
  __shared__ float sb[32];
  for (int i = threadIdx.x; i < 32 * 147; i += blockDim.x) sw[i] = w7f[i];
  if (threadIdx.x < 32) sb[threadIdx.x] = b7f[threadIdx.x];
  __syncthreads();
  int n = blockIdx.x * blockDim.x + threadIdx.x;   // token
  int b = n >> 14, rem = n & 16383, y = rem >> 7, x = rem & 127;
  float acc[32];
#pragma unroll
  for (int o = 0; o < 32; ++o) acc[o] = sb[o];
  const float* pb = pool + (size_t)b * 3 * LL;
  for (int ic = 0; ic < 3; ++ic)
    for (int ky = 0; ky < 7; ++ky) {
      int yy = y + ky - 3;
      if (yy < 0 || yy >= 128) continue;
      for (int kx = 0; kx < 7; ++kx) {
        int xx = x + kx - 3;
        if (xx < 0 || xx >= 128) continue;
        float in = pb[ic * LL + yy * 128 + xx];
        int widx = ic * 49 + ky * 7 + kx;
#pragma unroll
        for (int o = 0; o < 32; ++o) acc[o] += in * sw[o * 147 + widx];
      }
    }
  _Float16* dst = a1 + (size_t)n * 32;
#pragma unroll
  for (int o = 0; o < 32; ++o) dst[o] = (_Float16)fmaxf(acc[o], 0.f);
}

// ---------------- conv3x3 as 9-tap WMMA implicit GEMM (+BN fold, ReLU) ----------------
__global__ void __launch_bounds__(32 * GWAVES) k_conv3(
    const _Float16* __restrict__ a1, const _Float16* __restrict__ w3h,
    const float* __restrict__ bias, _Float16* __restrict__ a2) {
  int lane = threadIdx.x & 31;
  int wv = threadIdx.x >> 5;
  int hi = lane >> 4;
  int m0 = blockIdx.y * 16;
  v16h af[9];
#pragma unroll
  for (int t = 0; t < 9; ++t) af[t] = wmma_load_a(w3h + t * 1024, 32, m0, 0, lane);
  long tile0 = ((long)blockIdx.x * GWAVES + wv) * GTILES;
  for (int tt = 0; tt < GTILES; ++tt) {
    long n0 = (tile0 + tt) * 16;
    int b = (int)(n0 >> 14);
    int rem = (int)(n0 & 16383);
    int y = rem >> 7, x0 = rem & 127;
    v8f c = {};
#pragma unroll
    for (int dy = -1; dy <= 1; ++dy) {
      int yy = y + dy;
      if (yy < 0 || yy >= 128) continue;            // wave-uniform
#pragma unroll
      for (int dx = -1; dx <= 1; ++dx) {
        int xx = x0 + (lane & 15) + dx;
        v16h bf = {};
        if (xx >= 0 && xx < 128) {
          const _Float16* p = a1 + ((((long)b * 128 + yy) * 128 + xx) * 32 + 16 * hi);
          bf = *(const v16h*)p;
        }
        c = __builtin_amdgcn_wmma_f32_16x16x32_f16(false, af[(dy + 1) * 3 + (dx + 1)],
                                                   false, bf, (short)0, c, false, false);
      }
    }
    long n = n0 + (lane & 15);
#pragma unroll
    for (int r = 0; r < 8; ++r) {
      int m = m0 + r + 8 * hi;
      a2[n * 32 + m] = (_Float16)fmaxf(c[r] + bias[m], 0.f);
    }
  }
}

// ---------------- generic token GEMM Y = X @ W^T with epilogue modes ----------------
// mode 0: [bias][relu] -> Yf (+Yh);  mode 1: 0.5*acc + 0.5*res -> Yf + Yh
// mode 2: relu(acc+bias) + p1 + p2 -> outNCHW
__global__ void __launch_bounds__(32 * GWAVES) k_gemm(
    const _Float16* __restrict__ X, const _Float16* __restrict__ Wt,
    const float* __restrict__ bias, int ldK, int ksteps,
    float* __restrict__ Yf, _Float16* __restrict__ Yh, int Mld,
    const float* __restrict__ res, const float* __restrict__ p1,
    const float* __restrict__ p2, float* __restrict__ outN,
    int mode, int relu) {
  int lane = threadIdx.x & 31;
  int wv = threadIdx.x >> 5;
  int hi = lane >> 4;
  int m0 = blockIdx.y * 16;
  v16h a0 = wmma_load_a(Wt, ldK, m0, 0, lane);
  v16h a1 = {};
  if (ksteps > 1) a1 = wmma_load_a(Wt, ldK, m0, 32, lane);
  long tile0 = ((long)blockIdx.x * GWAVES + wv) * GTILES;
  for (int t = 0; t < GTILES; ++t) {
    long n0 = (tile0 + t) * 16;
    if (t + 1 < GTILES) {   // prefetch next tile's B rows (global_prefetch_b8)
      const _Float16* pf = X + (n0 + 16 + (lane & 15)) * (long)ldK;
      __builtin_prefetch(pf, 0, 1);
    }
    v8f c = {};
    v16h b0 = wmma_load_b(X, n0, ldK, 0, lane);
    c = __builtin_amdgcn_wmma_f32_16x16x32_f16(false, a0, false, b0, (short)0, c, false, false);
    if (ksteps > 1) {
      v16h b1 = wmma_load_b(X, n0, ldK, 32, lane);
      c = __builtin_amdgcn_wmma_f32_16x16x32_f16(false, a1, false, b1, (short)0, c, false, false);
    }
    long n = n0 + (lane & 15);
#pragma unroll
    for (int r = 0; r < 8; ++r) {
      int m = m0 + r + 8 * hi;
      float v = c[r];
      if (mode == 0) {
        if (bias) v += bias[m];
        if (relu) v = fmaxf(v, 0.f);
        Yf[n * Mld + m] = v;
        if (Yh) Yh[n * Mld + m] = (_Float16)v;
      } else if (mode == 1) {
        v = 0.5f * v + 0.5f * res[n * Mld + m];
        Yf[n * Mld + m] = v;
        Yh[n * Mld + m] = (_Float16)v;
      } else {
        v = fmaxf(v + bias[m], 0.f) + p1[n] + p2[n];
        long bb = n >> 14, ll = n & 16383;
        outN[(bb * 32 + m) * (long)LL + ll] = v;
      }
    }
  }
}

// ---------------- depthwise causal conv1d (k=4) + SiLU ----------------
__global__ void k_conv1d(const float* __restrict__ xz, const float* __restrict__ cw,
                         const float* __restrict__ cb, float* __restrict__ uf,
                         _Float16* __restrict__ uh) {
  int i = blockIdx.x * blockDim.x + threadIdx.x;   // over BL*64
  if (i >= BL * 64) return;
  int d = i & 63;
  int n = i >> 6;
  int l = n & 16383;
  float acc = cb[d];
#pragma unroll
  for (int t = 0; t < 4; ++t) {
    int lt = l - 3 + t;
    if (lt >= 0) acc += cw[d * 4 + t] * xz[(size_t)(n - 3 + t) * 128 + d];
  }
  float u = acc / (1.f + expf(-acc));
  uf[(size_t)n * 64 + d] = u;
  uh[(size_t)n * 64 + d] = (_Float16)u;
}

// -------- selective scan: 8 blocks x 1024 threads, (d,s) per thread --------
// Software-pipelined: step l+1 loads issued before step l compute; prefetch
// runs ~32 steps ahead so the pipelined loads hit WGP$/L2.
__global__ void __launch_bounds__(1024) k_scan(
    const float* __restrict__ xdbl, const float* __restrict__ uf,
    const float* __restrict__ xz, const float* __restrict__ dtw,
    const float* __restrict__ dtb, const float* __restrict__ Ac,
    const float* __restrict__ Dpv, _Float16* __restrict__ yg) {
  int b = blockIdx.x;
  int d = threadIdx.x >> 4, s = threadIdx.x & 15;
  float w0 = dtw[d * 2 + 0], w1 = dtw[d * 2 + 1], db = dtb[d];
  float A = Ac[d * 16 + s], Dd = Dpv[d];
  float h = 0.f;
  size_t nbase = (size_t)b * LL;

  // preload step 0
  float dt0, dt1, Bs, Cs, ud, zv;
  {
    const float* xr = xdbl + nbase * 48;
    dt0 = xr[0]; dt1 = xr[1]; Bs = xr[2 + s]; Cs = xr[18 + s];
    ud  = uf[nbase * 64 + d];
    zv  = (s == 0) ? xz[nbase * 128 + 64 + d] : 0.f;
  }
  for (int l = 0; l < LL; ++l) {
    size_t n = nbase + l;
    // look-ahead prefetch (a few threads cover the upcoming cachelines)
    if (s == 0 && (l & 7) == 0 && l + 32 < LL) {
      const char* pfa = (const char*)(xdbl + (n + 32) * 48);
      __builtin_prefetch(pfa + (d & 3) * 64, 0, 1);
      const char* pfb = (const char*)(uf + (n + 32) * 64);
      __builtin_prefetch(pfb + (d & 3) * 64, 0, 1);
    }
    float cdt0 = dt0, cdt1 = dt1, cBs = Bs, cCs = Cs, cud = ud, czv = zv;
    if (l + 1 < LL) {       // issue next step's independent loads now
      const float* xr = xdbl + (n + 1) * 48;
      dt0 = xr[0]; dt1 = xr[1]; Bs = xr[2 + s]; Cs = xr[18 + s];
      ud  = uf[(n + 1) * 64 + d];
      if (s == 0) zv = xz[(n + 1) * 128 + 64 + d];
    }
    float pre = cdt0 * w0 + cdt1 * w1 + db;
    float delta = fmaxf(pre, 0.f) + log1pf(expf(-fabsf(pre)));   // softplus
    float dA = expf(delta * A);
    h = dA * h + (delta * cud) * cBs;
    float c = h * cCs;
    c += __shfl_xor(c, 1, 32);
    c += __shfl_xor(c, 2, 32);
    c += __shfl_xor(c, 4, 32);
    c += __shfl_xor(c, 8, 32);
    if (s == 0) {
      float y = c + cud * Dd;
      yg[n * 64 + d] = (_Float16)(y * (czv / (1.f + expf(-czv))));
    }
  }
}

// ---------------- 1-channel conv + token LayerNorm (per batch) ----------------
__global__ void __launch_bounds__(1024) k_pln(
    const float* __restrict__ X, const float* __restrict__ pw, const float* __restrict__ pb,
    const float* __restrict__ lng, const float* __restrict__ lnb, float* __restrict__ pn) {
  __shared__ float wsum[32], wsq[32], bcast[2];
  int b = blockIdx.x, tid = threadIdx.x;
  float pbv = pb[0];
  float sv[16];
  float sum = 0.f, sq = 0.f;
#pragma unroll
  for (int i = 0; i < 16; ++i) {
    size_t n = (size_t)b * LL + tid + i * 1024;
    const float* xr = X + n * 32;
    float a = pbv;
#pragma unroll
    for (int k = 0; k < 32; ++k) a += xr[k] * pw[k];
    sv[i] = a; sum += a; sq += a * a;
  }
  for (int off = 16; off >= 1; off >>= 1) {
    sum += __shfl_xor(sum, off, 32);
    sq  += __shfl_xor(sq,  off, 32);
  }
  int lane = tid & 31, wv = tid >> 5;
  if (lane == 0) { wsum[wv] = sum; wsq[wv] = sq; }
  __syncthreads();
  if (wv == 0) {
    float s2 = wsum[lane], q2 = wsq[lane];
    for (int off = 16; off >= 1; off >>= 1) {
      s2 += __shfl_xor(s2, off, 32);
      q2 += __shfl_xor(q2, off, 32);
    }
    if (lane == 0) {
      float mu = s2 / (float)LL;
      float var = q2 / (float)LL - mu * mu;
      bcast[0] = mu;
      bcast[1] = rsqrtf(var + 1e-5f);
    }
  }
  __syncthreads();
  float mu = bcast[0], inv = bcast[1];
  float g = lng[0], be = lnb[0];
#pragma unroll
  for (int i = 0; i < 16; ++i) {
    size_t n = (size_t)b * LL + tid + i * 1024;
    pn[n] = (sv[i] - mu) * inv * g + be;
  }
}

// ---------------- launch ----------------
extern "C" void kernel_launch(void* const* d_in, const int* in_sizes, int n_in,
                              void* d_out, int out_size, void* d_ws, size_t ws_size,
                              hipStream_t stream) {
  (void)in_sizes; (void)n_in; (void)out_size; (void)ws_size;
  const float* x    = (const float*)d_in[0];
  const float* w7   = (const float*)d_in[1];
  const float* b7   = (const float*)d_in[2];
  const float* g1   = (const float*)d_in[3];
  const float* be1  = (const float*)d_in[4];
  const float* m1   = (const float*)d_in[5];
  const float* v1   = (const float*)d_in[6];
  const float* w3   = (const float*)d_in[7];
  const float* b3   = (const float*)d_in[8];
  const float* g2   = (const float*)d_in[9];
  const float* be2  = (const float*)d_in[10];
  const float* m2   = (const float*)d_in[11];
  const float* v2   = (const float*)d_in[12];
  const float* w1   = (const float*)d_in[13];
  const float* b1   = (const float*)d_in[14];
  const float* g3   = (const float*)d_in[15];
  const float* be3  = (const float*)d_in[16];
  const float* m3   = (const float*)d_in[17];
  const float* v3   = (const float*)d_in[18];
  const float* in_w = (const float*)d_in[19];
  const float* cw   = (const float*)d_in[20];
  const float* cb   = (const float*)d_in[21];
  const float* xp_w = (const float*)d_in[22];
  const float* dt_w = (const float*)d_in[23];
  const float* dt_b = (const float*)d_in[24];
  const float* A_log= (const float*)d_in[25];
  const float* Dp   = (const float*)d_in[26];
  const float* out_w= (const float*)d_in[27];
  const float* w4   = (const float*)d_in[28];
  const float* b4   = (const float*)d_in[29];
  const float* g4   = (const float*)d_in[30];
  const float* be4  = (const float*)d_in[31];
  const float* m4   = (const float*)d_in[32];
  const float* v4   = (const float*)d_in[33];
  const float* p1w  = (const float*)d_in[34];
  const float* p1b  = (const float*)d_in[35];
  const float* p2w  = (const float*)d_in[36];
  const float* p2b  = (const float*)d_in[37];
  const float* lng  = (const float*)d_in[38];
  const float* lnb  = (const float*)d_in[39];

  char* ws = (char*)d_ws;
  float*     pool = (float*)(ws + OFF_POOL);
  _Float16*  a1   = (_Float16*)(ws + OFF_A1);
  _Float16*  a2   = (_Float16*)(ws + OFF_A2);
  float*     x3f  = (float*)(ws + OFF_X3F);
  _Float16*  x3h  = (_Float16*)(ws + OFF_X3H);
  float*     xz   = (float*)(ws + OFF_XZ);
  _Float16*  u16  = (_Float16*)(ws + OFF_U16);
  float*     uf   = (float*)(ws + OFF_UF);
  float*     xdbl = (float*)(ws + OFF_XDBL);
  _Float16*  yg   = (_Float16*)(ws + OFF_YG);
  float*     xnf  = (float*)(ws + OFF_XNF);
  _Float16*  xnh  = (_Float16*)(ws + OFF_XNH);
  float*     p1n  = (float*)(ws + OFF_P1N);
  float*     p2n  = (float*)(ws + OFF_P2N);

  char* wb = ws + OFF_W;
  float*    w7f  = (float*)(wb + OW_W7);
  float*    b7f  = (float*)(wb + OW_B7);
  _Float16* w3h  = (_Float16*)(wb + OW_W3);
  float*    b3f  = (float*)(wb + OW_B3);
  _Float16* w1h  = (_Float16*)(wb + OW_W1);
  float*    b1f  = (float*)(wb + OW_B1);
  _Float16* w4h  = (_Float16*)(wb + OW_W4);
  float*    b4f  = (float*)(wb + OW_B4);
  _Float16* inwh = (_Float16*)(wb + OW_INW);
  _Float16* xpwh = (_Float16*)(wb + OW_XPW);
  _Float16* outwh= (_Float16*)(wb + OW_OUTW);
  float*    Ac   = (float*)(wb + OW_AC);

  const int gx = BL / (16 * GTILES * GWAVES);   // 256 blocks of 4 waves

  // 1. fold BN, convert weights
  k_prep<<<32, 256, 0, stream>>>(w7, b7, g1, be1, m1, v1, w3, b3, g2, be2, m2, v2,
                                 w1, b1, g3, be3, m3, v3, in_w, xp_w, out_w, A_log,
                                 w4, b4, g4, be4, m4, v4,
                                 w7f, b7f, w3h, b3f, w1h, b1f, w4h, b4f,
                                 inwh, xpwh, outwh, Ac);
  // 2. maxpool
  k_pool<<<(BATCH * 3 * LL) / 256, 256, 0, stream>>>(x, pool);
  // 3. conv7x7 + BN + ReLU -> a1 (channel-last f16)
  k_conv7<<<BL / 256, 256, 0, stream>>>(pool, w7f, b7f, a1);
  // 4. conv3x3 via 9-tap WMMA -> a2
  k_conv3<<<dim3(gx, 2), 32 * GWAVES, 0, stream>>>(a1, w3h, b3f, a2);
  // 5. conv1x1 cbr (K=32,M=32) -> x3 f32+f16
  k_gemm<<<dim3(gx, 2), 32 * GWAVES, 0, stream>>>(a2, w1h, b1f, 32, 1,
        x3f, x3h, 32, nullptr, nullptr, nullptr, nullptr, 0, 1);
  // 6. p1 = LN(conv1(x3))
  k_pln<<<BATCH, 1024, 0, stream>>>(x3f, p1w, p1b, lng, lnb, p1n);
  // 7. in_proj (K=32,M=128) -> xz f32
  k_gemm<<<dim3(gx, 8), 32 * GWAVES, 0, stream>>>(x3h, inwh, nullptr, 32, 1,
        xz, nullptr, 128, nullptr, nullptr, nullptr, nullptr, 0, 0);
  // 8. depthwise causal conv1d + SiLU -> u
  k_conv1d<<<(BL * 64) / 256, 256, 0, stream>>>(xz, cw, cb, uf, u16);
  // 9. x_proj (K=64,M=48 padded) -> xdbl f32
  k_gemm<<<dim3(gx, 3), 32 * GWAVES, 0, stream>>>(u16, xpwh, nullptr, 64, 2,
        xdbl, nullptr, 48, nullptr, nullptr, nullptr, nullptr, 0, 0);
  // 10. selective scan + gate -> yg f16
  k_scan<<<BATCH, 1024, 0, stream>>>(xdbl, uf, xz, dt_w, dt_b, Ac, Dp, yg);
  // 11. out_proj (K=64,M=32) + residual mix -> xnew f32+f16
  k_gemm<<<dim3(gx, 2), 32 * GWAVES, 0, stream>>>(yg, outwh, nullptr, 64, 2,
        xnf, xnh, 32, x3f, nullptr, nullptr, nullptr, 1, 0);
  // 12. p2 = LN(conv1(xnew))
  k_pln<<<BATCH, 1024, 0, stream>>>(xnf, p2w, p2b, lng, lnb, p2n);
  // 13. final conv1x1 cbr + p1 + p2 -> d_out (NCHW f32)
  k_gemm<<<dim3(gx, 2), 32 * GWAVES, 0, stream>>>(xnh, w4h, b4f, 32, 1,
        nullptr, nullptr, 32, nullptr, p1n, p2n, (float*)d_out, 2, 1);
}
